// A2CDense_30494267801740
// MI455X (gfx1250) — compile-verified, matrix operand order checked by reading, not run on hardware
//
#include <hip/hip_runtime.h>
#include <hip/hip_bf16.h>

// ---------------------------------------------------------------------------
// GraphNet block for MI455X (gfx1250, wave32, WMMA).
//   B=64, NV=512, NE=2048, all feature widths 32, MLPs 128/96 -> 32x3 -> 32.
// Strategy:
//   * recover s_idx/r_idx from dense one-hot incidence (the only BW-heavy op)
//     with b128 loads + v-dim split across blocks (atomicMax merge)
//   * run every Linear as v_wmma_f32_16x16x32_f16 (16 rows/wave, 2 N-tiles)
//   * f16 A/B fragments, f32 accumulate, exact-erf GELU, LDS inter-layer
//     transpose (wave-local, LDS in-order => no barriers)
//   * scatter agg / edge+node sums with global f32 atomics
// ---------------------------------------------------------------------------

typedef _Float16 h16;
typedef __attribute__((ext_vector_type(16))) _Float16 v16h;
typedef __attribute__((ext_vector_type(8)))  float    v8f;

#define NB 64
#define NVN 512
#define NEE 2048
#define ACT_LD 34   // padded 16x32 activation tile row stride (halves)

struct MLPW { const float* W[4]; const float* b[4]; };

__device__ __forceinline__ float gelu_exact(float x) {
  return 0.5f * x * (1.0f + erff(x * 0.70710678118654752f));
}

// 16-halves-per-lane fragment per CDNA5 ISA 7.12.2 (16-bit A matrix 16x32):
// VGPR v holds K pair {k,k+1}, k = 2*(v&3) + (v>=4?16:0) + (lane>=16?8:0).
// Same mapping reused for B columns: B(k,n) = W[n][k], p = &W[n][0].
__device__ __forceinline__ v16h frag16(const h16* p, int sub) {
  v16h a;
#pragma unroll
  for (int v = 0; v < 8; ++v) {
    int k = 2 * (v & 3) + ((v & 4) ? 16 : 0) + (sub ? 8 : 0);
    a[2 * v]     = p[k];
    a[2 * v + 1] = p[k + 1];
  }
  return a;
}

// One wave: 16 rows through a 4-layer MLP (out width 32, GELU incl. last).
// LDS weights layout (f16, flat [dout][din]): layer0 at 0 (32*K0),
// layer l>=1 at 32*K0 + (l-1)*1024. bias: 4*32 f32.
template <int K0>
__device__ void mlp4_wave(const h16* feat, int fstride, const h16* wts,
                          const float* bias, h16* act, float* dout, int lane) {
  const int sub = lane >> 4;
  const int row = lane & 15;
  float d[16];
  for (int layer = 0; layer < 4; ++layer) {
    const int K = (layer == 0) ? K0 : 32;
    const h16* src = (layer == 0) ? (feat + row * fstride) : (act + row * ACT_LD);
    const h16* wb  = (layer == 0) ? wts : (wts + 32 * K0 + (layer - 1) * 1024);
    const float* bl = bias + layer * 32;
    v8f c0 = {}; v8f c1 = {};
    for (int ks = 0; ks < K / 32; ++ks) {
      v16h a  = frag16(src + ks * 32, sub);
      v16h b0 = frag16(wb + row * K + ks * 32, sub);
      v16h b1 = frag16(wb + (row + 16) * K + ks * 32, sub);
      c0 = __builtin_amdgcn_wmma_f32_16x16x32_f16(false, a, false, b0,
                                                  (short)0, c0, false, false);
      c1 = __builtin_amdgcn_wmma_f32_16x16x32_f16(false, a, false, b1,
                                                  (short)0, c1, false, false);
    }
    const float bn0 = bl[row];
    const float bn1 = bl[row + 16];
#pragma unroll
    for (int i = 0; i < 8; ++i) {
      d[i]     = gelu_exact(c0[i] + bn0);
      d[8 + i] = gelu_exact(c1[i] + bn1);
    }
    if (layer < 3) {
      // D element i sits at (M=i+8*sub, N=row[+16]); transpose to row-major
      // activation tile via wave-local LDS (DS ops in-order within a wave).
#pragma unroll
      for (int i = 0; i < 8; ++i) {
        act[(i + 8 * sub) * ACT_LD + row]      = (h16)d[i];
        act[(i + 8 * sub) * ACT_LD + row + 16] = (h16)d[8 + i];
      }
    }
  }
#pragma unroll
  for (int i = 0; i < 16; ++i) dout[i] = d[i];
}

// ---------------------------------------------------------------------------
// Kernel 0: zero scratch region (sidx + ridx + agg + sumE + sumV).
__global__ __launch_bounds__(256) void zero_kernel(float* p, int n) {
  int i = blockIdx.x * 256 + threadIdx.x;
  if (i < n) p[i] = 0.0f;
}

// ---------------------------------------------------------------------------
// Kernel 1: recover integer indices from dense one-hot incidence matrices.
// This is the HBM-bound phase (537 MB): b128 loads (4 edges / thread), and the
// 512-deep v scan is split into 4 chunks of 128 across blocks for 4x more
// memory-level parallelism. Per-chunk argmax candidates merge via atomicMax
// into zero-initialized buffers (one-hot => exactly one chunk finds the 1;
// index 0 is the correct default).  global_prefetch_b8 lookahead per row.
__global__ __launch_bounds__(256) void index_kernel(const float4* __restrict__ Es,
                                                    const float4* __restrict__ Er,
                                                    int* __restrict__ sidx,
                                                    int* __restrict__ ridx) {
  const int blk = blockIdx.x;
  const int vc  = blk & 3;                   // v-chunk: 128 rows each
  const int et  = (blk >> 2) & 1;            // edge tile: 1024 edges each
  const int b   = blk >> 3;
  const int e4  = et * 256 + threadIdx.x;    // float4 column (NE/4 = 512/row)
  const size_t rowq = NEE / 4;
  const float4* ps = Es + (size_t)b * NVN * rowq + e4;
  const float4* pr = Er + (size_t)b * NVN * rowq + e4;
  int s0 = 0, s1 = 0, s2 = 0, s3 = 0;
  int r0 = 0, r1 = 0, r2 = 0, r3 = 0;
  const int vbase = vc * 128;
  for (int i = 0; i < 128; ++i) {
    const int v = vbase + i;
    if (v + 4 < NVN) {
      __builtin_prefetch(ps + (size_t)(v + 4) * rowq, 0, 1);
      __builtin_prefetch(pr + (size_t)(v + 4) * rowq, 0, 1);
    }
    const float4 a = ps[(size_t)v * rowq];
    const float4 c = pr[(size_t)v * rowq];
    if (a.x > 0.5f) s0 = v;
    if (a.y > 0.5f) s1 = v;
    if (a.z > 0.5f) s2 = v;
    if (a.w > 0.5f) s3 = v;
    if (c.x > 0.5f) r0 = v;
    if (c.y > 0.5f) r1 = v;
    if (c.z > 0.5f) r2 = v;
    if (c.w > 0.5f) r3 = v;
  }
  const int eb = b * NEE + e4 * 4;
  if (s0) atomicMax(&sidx[eb + 0], s0);
  if (s1) atomicMax(&sidx[eb + 1], s1);
  if (s2) atomicMax(&sidx[eb + 2], s2);
  if (s3) atomicMax(&sidx[eb + 3], s3);
  if (r0) atomicMax(&ridx[eb + 0], r0);
  if (r1) atomicMax(&ridx[eb + 1], r1);
  if (r2) atomicMax(&ridx[eb + 2], r2);
  if (r3) atomicMax(&ridx[eb + 3], r3);
}

// ---------------------------------------------------------------------------
// Kernel 2: edge MLP. One block = (batch b, 128-edge tile); 8 waves x 16 edges.
__global__ __launch_bounds__(256) void edge_kernel(
    const float* __restrict__ Ea, const float* __restrict__ Va,
    const float* __restrict__ u, const int* __restrict__ sidx,
    const int* __restrict__ ridx, MLPW w, float* __restrict__ outE,
    float* __restrict__ agg, float* __restrict__ sumE) {
  constexpr int FS = 132;                        // padded 128-feat row stride
  __shared__ h16  feat[128 * FS];
  __shared__ h16  wts[32 * 128 + 3 * 1024];      // 7168 halves
  __shared__ float bias[128];
  __shared__ h16  act[8 * 16 * ACT_LD];
  __shared__ float outT[32 * 132];
  __shared__ int  sI[128], rI[128];

  const int b   = blockIdx.x >> 4;               // NE/128 = 16 tiles
  const int e0  = (blockIdx.x & 15) << 7;
  const int tid = threadIdx.x;

  if (tid < 128) {
    sI[tid] = sidx[b * NEE + e0 + tid];
    rI[tid] = ridx[b * NEE + e0 + tid];
  }
  for (int i = tid; i < 7168; i += 256) {
    float v = (i < 4096) ? w.W[0][i] : w.W[1 + ((i - 4096) >> 10)][(i - 4096) & 1023];
    wts[i] = (h16)v;
  }
  if (tid < 128) bias[tid] = w.b[tid >> 5][tid & 31];
  __syncthreads();

  // feature assembly: [E_a | V_a[s] | V_a[r] | u] -> f16 LDS, coalesced in e
  for (int it = 0; it < 16; ++it) {
    int idx = it * 256 + tid;
    int d = idx >> 7, e = idx & 127;
    feat[e * FS + d]      = (h16)Ea[((size_t)b * 32 + d) * NEE + e0 + e];
    feat[e * FS + 32 + d] = (h16)Va[((size_t)b * 32 + d) * NVN + sI[e]];
    feat[e * FS + 64 + d] = (h16)Va[((size_t)b * 32 + d) * NVN + rI[e]];
    feat[e * FS + 96 + d] = (h16)u[b * 32 + d];
  }
  __syncthreads();

  const int wv = tid >> 5, lane = tid & 31;
  float dout[16];
  mlp4_wave<128>(&feat[(wv * 16) * FS], FS, wts, bias,
                 &act[wv * 16 * ACT_LD], dout, lane);
  {
    const int sub = lane >> 4, n0 = lane & 15;
#pragma unroll
    for (int i = 0; i < 8; ++i) {
      outT[n0 * 132 + wv * 16 + i + 8 * sub]        = dout[i];
      outT[(n0 + 16) * 132 + wv * 16 + i + 8 * sub] = dout[8 + i];
    }
  }
  __syncthreads();

  // coalesced out_E store + receiver scatter (global f32 atomics)
  for (int it = 0; it < 16; ++it) {
    int idx = it * 256 + tid;
    int d = idx >> 7, e = idx & 127;
    float v = outT[d * 132 + e];
    outE[((size_t)b * 32 + d) * NEE + e0 + e] = v;
    atomicAdd(&agg[((size_t)b * 32 + d) * NVN + rI[e]], v);
  }
  if (tid < 32) {
    float s = 0.0f;
    for (int e = 0; e < 128; ++e) s += outT[tid * 132 + e];
    atomicAdd(&sumE[b * 32 + tid], s);
  }
}

// ---------------------------------------------------------------------------
// Kernel 3: node MLP. One block = (batch b, 128-node tile); K0 = 96.
__global__ __launch_bounds__(256) void node_kernel(
    const float* __restrict__ Va, const float* __restrict__ agg,
    const float* __restrict__ u, MLPW w, float* __restrict__ outV,
    float* __restrict__ sumV) {
  constexpr int FS = 100;                        // padded 96-feat row stride
  __shared__ h16  feat[128 * FS];
  __shared__ h16  wts[32 * 96 + 3 * 1024];       // 6144 halves
  __shared__ float bias[128];
  __shared__ h16  act[8 * 16 * ACT_LD];
  __shared__ float outT[32 * 132];

  const int b   = blockIdx.x >> 2;               // NV/128 = 4 tiles
  const int v0  = (blockIdx.x & 3) << 7;
  const int tid = threadIdx.x;

  for (int i = tid; i < 6144; i += 256) {
    float v = (i < 3072) ? w.W[0][i] : w.W[1 + ((i - 3072) >> 10)][(i - 3072) & 1023];
    wts[i] = (h16)v;
  }
  if (tid < 128) bias[tid] = w.b[tid >> 5][tid & 31];
  __syncthreads();

  for (int it = 0; it < 12; ++it) {              // 24 feats (d < 24) * 128 nodes / 256
    int idx = it * 256 + tid;
    int d = idx >> 7, e = idx & 127;
    float v;
    if (d < 32)      v = Va[((size_t)b * 32 + d) * NVN + v0 + e];
    else if (d < 64) v = agg[((size_t)b * 32 + (d - 32)) * NVN + v0 + e];
    else             v = u[b * 32 + (d - 64)];
    feat[e * FS + d] = (h16)v;
  }
  __syncthreads();

  const int wv = tid >> 5, lane = tid & 31;
  float dout[16];
  mlp4_wave<96>(&feat[(wv * 16) * FS], FS, wts, bias,
                &act[wv * 16 * ACT_LD], dout, lane);
  {
    const int sub = lane >> 4, n0 = lane & 15;
#pragma unroll
    for (int i = 0; i < 8; ++i) {
      outT[n0 * 132 + wv * 16 + i + 8 * sub]        = dout[i];
      outT[(n0 + 16) * 132 + wv * 16 + i + 8 * sub] = dout[8 + i];
    }
  }
  __syncthreads();

  for (int it = 0; it < 16; ++it) {
    int idx = it * 256 + tid;
    int d = idx >> 7, e = idx & 127;
    outV[((size_t)b * 32 + d) * NVN + v0 + e] = outT[d * 132 + e];
  }
  if (tid < 32) {
    float s = 0.0f;
    for (int e = 0; e < 128; ++e) s += outT[tid * 132 + e];
    atomicAdd(&sumV[b * 32 + tid], s);
  }
}

// ---------------------------------------------------------------------------
// Kernel 4: global MLP. One block, 4 waves x 16 batch rows = 64 rows. K0 = 96.
__global__ __launch_bounds__(128) void global_kernel(
    const float* __restrict__ u, const float* __restrict__ sumV,
    const float* __restrict__ sumE, MLPW w, float* __restrict__ outU) {
  constexpr int FS = 100;
  __shared__ h16  feat[64 * FS];
  __shared__ h16  wts[32 * 96 + 3 * 1024];
  __shared__ float bias[128];
  __shared__ h16  act[4 * 16 * ACT_LD];

  const int tid = threadIdx.x;
  for (int i = tid; i < 6144; i += 128) {
    float v = (i < 3072) ? w.W[0][i] : w.W[1 + ((i - 3072) >> 10)][(i - 3072) & 1023];
    wts[i] = (h16)v;
  }
  if (tid < 128) bias[tid] = w.b[tid >> 5][tid & 31];
  for (int i = tid; i < 64 * 96; i += 128) {
    int row = i / 96, f = i % 96;
    float v;
    if (f < 32)      v = u[row * 32 + f];
    else if (f < 64) v = sumV[row * 32 + (f - 32)];
    else             v = sumE[row * 32 + (f - 64)];
    feat[row * FS + f] = (h16)v;
  }
  __syncthreads();

  const int wv = tid >> 5, lane = tid & 31;
  float dout[16];
  mlp4_wave<96>(&feat[(wv * 16) * FS], FS, wts, bias,
                &act[wv * 16 * ACT_LD], dout, lane);
  const int sub = lane >> 4, n0 = lane & 15;
#pragma unroll
  for (int i = 0; i < 8; ++i) {
    int bb = wv * 16 + i + 8 * sub;
    outU[bb * 32 + n0]      = dout[i];
    outU[bb * 32 + n0 + 16] = dout[8 + i];
  }
}

// ---------------------------------------------------------------------------
extern "C" void kernel_launch(void* const* d_in, const int* in_sizes, int n_in,
                              void* d_out, int out_size, void* d_ws, size_t ws_size,
                              hipStream_t stream) {
  const float* Ea = (const float*)d_in[0];
  const float* Es = (const float*)d_in[1];
  const float* Er = (const float*)d_in[2];
  const float* Va = (const float*)d_in[3];
  const float* u  = (const float*)d_in[4];
  MLPW fe, fv, fu;
  for (int i = 0; i < 4; ++i) {
    fe.W[i] = (const float*)d_in[5 + i];   fe.b[i] = (const float*)d_in[9 + i];
    fv.W[i] = (const float*)d_in[13 + i];  fv.b[i] = (const float*)d_in[17 + i];
    fu.W[i] = (const float*)d_in[21 + i];  fu.b[i] = (const float*)d_in[25 + i];
  }

  // workspace layout (contiguous so one zero pass covers everything)
  char* ws = (char*)d_ws;
  int*   sidx = (int*)ws;                                   // 64*2048
  int*   ridx = (int*)(ws + (size_t)NB * NEE * 4);          // 64*2048
  float* agg  = (float*)(ws + (size_t)2 * NB * NEE * 4);    // 64*32*512
  float* sumE = agg + (size_t)NB * 32 * NVN;                // 64*32
  float* sumV = sumE + NB * 32;                             // 64*32

  float* outE = (float*)d_out;                              // [64,32,2048]
  float* outV = outE + (size_t)NB * 32 * NEE;               // [64,32,512]
  float* outU = outV + (size_t)NB * 32 * NVN;               // [64,32]

  const int nz = 2 * NB * NEE + NB * 32 * NVN + 2 * NB * 32;
  zero_kernel<<<(nz + 255) / 256, 256, 0, stream>>>((float*)ws, nz);
  index_kernel<<<NB * 8, 256, 0, stream>>>((const float4*)Es, (const float4*)Er,
                                           sidx, ridx);
  edge_kernel<<<NB * (NEE / 128), 256, 0, stream>>>(Ea, Va, u, sidx, ridx, fe,
                                                    outE, agg, sumE);
  node_kernel<<<NB * (NVN / 128), 256, 0, stream>>>(Va, agg, u, fv, outV, sumV);
  global_kernel<<<1, 128, 0, stream>>>(u, sumV, sumE, fu, outU);
}
